// Encoder_90847148245060
// MI455X (gfx1250) — matrix operand order repaired and checked by reading, not compiled, not run on hardware
//
#include <hip/hip_runtime.h>
#include <hip/hip_bf16.h>
#include <stdint.h>

// ---------------------------------------------------------------------------
// Types for CDNA5 WMMA (wave32): 16x16x32 bf16 -> f32
// ---------------------------------------------------------------------------
typedef __attribute__((ext_vector_type(16))) __bf16 v16bf;
typedef __attribute__((ext_vector_type(8)))  float  v8f;

struct __align__(16) BF16x8 { __bf16 h[8]; };

// ---------------------------------------------------------------------------
// Async global->LDS copy (CDNA5 ASYNCcnt path), with safe fallbacks.
// Builtin params are v4i pointers in AS1 (global) / AS3 (LDS).
// ---------------------------------------------------------------------------
typedef int v4i_vs __attribute__((vector_size(16)));
typedef __attribute__((address_space(1))) v4i_vs* g_v4i_ptr;
typedef __attribute__((address_space(3))) v4i_vs* l_v4i_ptr;

__device__ __forceinline__ void async_copy16(const __bf16* gsrc, __bf16* ldst) {
#if defined(__AMDGCN__) && __has_builtin(__builtin_amdgcn_global_load_async_to_lds_b128)
  __builtin_amdgcn_global_load_async_to_lds_b128(
      (g_v4i_ptr)(__attribute__((address_space(1))) void*)(void*)gsrc,
      (l_v4i_ptr)(__attribute__((address_space(3))) void*)(void*)ldst,
      0, 0);
#else
  *reinterpret_cast<BF16x8*>(ldst) = *reinterpret_cast<const BF16x8*>(gsrc);
#endif
}

__device__ __forceinline__ void wait_async_lds() {
#if defined(__AMDGCN__) && __has_builtin(__builtin_amdgcn_s_wait_asynccnt)
  __builtin_amdgcn_s_wait_asynccnt(0);
#elif defined(__AMDGCN__)
  asm volatile("s_wait_asynccnt 0" ::: "memory");
#endif
}

// ---------------------------------------------------------------------------
// GEMM: C[M,N] = A[M,K](bf16, row-major) * B^T where B is stored [N,K] bf16.
// grid = (N/64, M/128), block = 256 (8 wave32).
// Each wave computes a 16(M) x 64(N) strip: 4 WMMA accumulators sharing one
// A fragment per k-step  ->  4 v_wmma per 2 global b128 loads.
// B is staged in 64-row x 256-k panels into LDS via async DMA (32 KB).
// Requires: M%128==0, N%64==0, K%256==0.
// ---------------------------------------------------------------------------
#define KP 256

__global__ __launch_bounds__(256)
void k_gemm_bf16_nt_wmma(const __bf16* __restrict__ A,
                         const __bf16* __restrict__ B,
                         float* __restrict__ C,
                         int M, int Nc, int K) {
  __shared__ __align__(16) __bf16 sB[64 * KP];   // 32 KB

  const int lane  = threadIdx.x & 31;
  const int wave  = threadIdx.x >> 5;
  const int jBase = blockIdx.x * 64;
  const int iBase = (blockIdx.y * 8 + wave) * 16;

  const int m    = lane & 15;      // A row within tile / B col within tile
  const int hi   = lane >> 4;      // lane group 0..15 vs 16..31
  const int aOff = hi ? 8 : 0;     // A: elems 0..7 at k0, 8..15 at k0+16
  const int bOff = hi ? 8 : 0;     // B: elems 0..7 at k0, 8..15 at k0+16

  const __bf16* Arow = A + (size_t)(iBase + m) * K;

  v8f acc[4] = {{}, {}, {}, {}};

  const int nPanels = K / KP;
  for (int p = 0; p < nPanels; ++p) {
    __syncthreads();   // previous panel fully consumed before overwrite

    // ---- stage 64 x KP B panel (rows jBase..jBase+63) via async DMA ----
    const __bf16* Bpan = B + (size_t)jBase * K + p * KP;
    for (int t = threadIdx.x; t < (64 * KP) / 8; t += 256) {
      const int r  = t / (KP / 8);
      const int c8 = (t % (KP / 8)) * 8;
      async_copy16(Bpan + (size_t)r * K + c8, sB + r * KP + c8);
    }
    wait_async_lds();
    __syncthreads();

    // ---- main WMMA loop over this panel ----
    const __bf16* Apan = Arow + p * KP;
    for (int kb = 0; kb < KP; kb += 32) {
      if (kb + 64 < KP) __builtin_prefetch(Apan + kb + 64, 0, 1);
      v16bf a;
      reinterpret_cast<BF16x8*>(&a)[0] = *reinterpret_cast<const BF16x8*>(Apan + kb + aOff);
      reinterpret_cast<BF16x8*>(&a)[1] = *reinterpret_cast<const BF16x8*>(Apan + kb + aOff + 16);
#pragma unroll
      for (int t = 0; t < 4; ++t) {
        const __bf16* Brow = sB + (size_t)(t * 16 + m) * KP + kb + bOff;
        v16bf b;
        reinterpret_cast<BF16x8*>(&b)[0] = *reinterpret_cast<const BF16x8*>(Brow);
        reinterpret_cast<BF16x8*>(&b)[1] = *reinterpret_cast<const BF16x8*>(Brow + 16);
        acc[t] = __builtin_amdgcn_wmma_f32_16x16x32_bf16(false, a, false, b,
                                                         (short)0, acc[t], false, false);
      }
    }
  }

  // ---- store: VGPR v holds row m=v(+8 for hi lanes), col = lane&15 ----
#pragma unroll
  for (int t = 0; t < 4; ++t) {
#pragma unroll
    for (int v = 0; v < 8; ++v) {
      const int mv = v + (hi ? 8 : 0);
      C[(size_t)(iBase + mv) * Nc + jBase + t * 16 + m] = acc[t][v];
    }
  }
}

// ---------------------------------------------------------------------------
// Elementwise / graph kernels
// ---------------------------------------------------------------------------
__global__ void k_fill(float* __restrict__ p, int n, float v) {
  int i = blockIdx.x * blockDim.x + threadIdx.x;
  if (i < n) p[i] = v;
}

__global__ void k_cast_bf16(const float* __restrict__ x, __bf16* __restrict__ y, int n) {
  int i = blockIdx.x * blockDim.x + threadIdx.x;
  if (i < n) y[i] = (__bf16)x[i];
}

// W[Kd,Nd] f32 -> Wt[Nd,Kd] bf16
__global__ void k_transpose_cast(const float* __restrict__ W, __bf16* __restrict__ Wt,
                                 int Kd, int Nd) {
  int i = blockIdx.x * blockDim.x + threadIdx.x;
  if (i < Kd * Nd) {
    int k = i / Nd, nn = i % Nd;
    Wt[(size_t)nn * Kd + k] = (__bf16)W[i];
  }
}

__global__ void k_deg_accum(const int* __restrict__ col, float* __restrict__ deg, int e) {
  int i = blockIdx.x * blockDim.x + threadIdx.x;
  if (i < e) atomicAdd(&deg[col[i]], 1.0f);
}

__global__ void k_dinv(const float* __restrict__ deg, float* __restrict__ dinv, int n) {
  int i = blockIdx.x * blockDim.x + threadIdx.x;
  if (i < n) { float d = deg[i]; dinv[i] = d > 0.f ? rsqrtf(d) : 0.f; }
}

// out[col] += xw[row] * dinv[row]*dinv[col]   (one block per edge)
__global__ void k_scatter_edges(const float* __restrict__ xw,
                                const int* __restrict__ row, const int* __restrict__ col,
                                const float* __restrict__ dinv,
                                float* __restrict__ acc, int C) {
  const int e = blockIdx.x;
  const int r = row[e], c = col[e];
  const float nrm = dinv[r] * dinv[c];
  const float* src = xw + (size_t)r * C;
  float* dst = acc + (size_t)c * C;
  for (int ch = threadIdx.x; ch < C; ch += blockDim.x)
    atomicAdd(&dst[ch], src[ch] * nrm);
}

// x = act( acc + xw * dinv^2 (self-loop) + bias )
__global__ void k_finalize(const float* __restrict__ acc, const float* __restrict__ xw,
                           const float* __restrict__ dinv, const float* __restrict__ bias,
                           float* __restrict__ out, int n, int C, int relu) {
  int i = blockIdx.x * blockDim.x + threadIdx.x;
  if (i < n * C) {
    int node = i / C, ch = i % C;
    float d = dinv[node];
    float v = acc[i] + xw[i] * d * d + bias[ch];
    out[i] = relu ? fmaxf(v, 0.f) : v;
  }
}

// t3[node] = dot(x1[node,:], W3[:,0])  (block per node)
__global__ __launch_bounds__(256)
void k_dot_w3(const float* __restrict__ x1, const float* __restrict__ W3,
              float* __restrict__ t3, int C) {
  __shared__ float red[256];
  const int node = blockIdx.x;
  const float* xr = x1 + (size_t)node * C;
  float s = 0.f;
  for (int ch = threadIdx.x; ch < C; ch += 256) s += xr[ch] * W3[ch];
  red[threadIdx.x] = s; __syncthreads();
  for (int st = 128; st; st >>= 1) {
    if (threadIdx.x < st) red[threadIdx.x] += red[threadIdx.x + st];
    __syncthreads();
  }
  if (threadIdx.x == 0) t3[node] = red[0];
}

__global__ void k_scatter_scalar(const float* __restrict__ t3,
                                 const int* __restrict__ row, const int* __restrict__ col,
                                 const float* __restrict__ dinv,
                                 float* __restrict__ pacc, int e) {
  int i = blockIdx.x * blockDim.x + threadIdx.x;
  if (i < e) {
    int r = row[i], c = col[i];
    atomicAdd(&pacc[c], t3[r] * dinv[r] * dinv[c]);
  }
}

__global__ void k_pos_final(const float* __restrict__ pacc, const float* __restrict__ t3,
                            const float* __restrict__ dinv, const float* __restrict__ b3,
                            float* __restrict__ posw, int n) {
  int i = blockIdx.x * blockDim.x + threadIdx.x;
  if (i < n) {
    float d = dinv[i];
    float v = pacc[i] + t3[i] * d * d + b3[0];
    posw[i] = 1.f / (1.f + __expf(-v));
  }
}

// in-place per-row: softmax(row)*pos (pos cancels; kept for fidelity), then sigmoid
__global__ __launch_bounds__(256)
void k_softmax_sigmoid(float* __restrict__ outm, const float* __restrict__ posw, int n) {
  __shared__ float red[256];
  const int rowi = blockIdx.x;
  float* p = outm + (size_t)rowi * n;
  const int tid = threadIdx.x;

  float m = -3.402823466e38f;
  for (int j = tid; j < n; j += 256) m = fmaxf(m, p[j]);
  red[tid] = m; __syncthreads();
  for (int st = 128; st; st >>= 1) {
    if (tid < st) red[tid] = fmaxf(red[tid], red[tid + st]);
    __syncthreads();
  }
  m = red[0]; __syncthreads();

  const float pw = posw[rowi];
  float s = 0.f;
  for (int j = tid; j < n; j += 256) {
    float v = __expf(p[j] - m) * pw;
    p[j] = v; s += v;
  }
  red[tid] = s; __syncthreads();
  for (int st = 128; st; st >>= 1) {
    if (tid < st) red[tid] += red[tid + st];
    __syncthreads();
  }
  const float inv = 1.f / red[0];
  __syncthreads();
  for (int j = tid; j < n; j += 256) {
    float pr = p[j] * inv;
    p[j] = 1.f / (1.f + __expf(-pr));
  }
}

// ---------------------------------------------------------------------------
// Orchestration
// ---------------------------------------------------------------------------
extern "C" void kernel_launch(void* const* d_in, const int* in_sizes, int n_in,
                              void* d_out, int out_size, void* d_ws, size_t ws_size,
                              hipStream_t stream) {
  const float* feat = (const float*)d_in[0];
  const int*   ei   = (const int*)d_in[1];   // [2,E] int32
  const float* W1   = (const float*)d_in[2]; // [512,512]
  const float* b1   = (const float*)d_in[3];
  const float* W2   = (const float*)d_in[4]; // [512,256]
  const float* b2   = (const float*)d_in[5];
  const float* W3   = (const float*)d_in[6]; // [512,1]
  const float* b3   = (const float*)d_in[7];

  const int INC = 512, HID = 512, OUTC = 256;
  const int n = in_sizes[0] / INC;   // 8192
  const int e = in_sizes[1] / 2;     // 131072
  const int* rowp = ei;
  const int* colp = ei + e;

  // workspace carve-out (256B aligned slices)
  char* w = (char*)d_ws; size_t off = 0;
  auto alloc = [&](size_t bytes) -> void* {
    void* p = w + off; off = (off + bytes + 255) & ~(size_t)255; return p;
  };
  float*  deg    = (float*)alloc((size_t)n * 4);
  float*  dinv   = (float*)alloc((size_t)n * 4);
  float*  t3     = (float*)alloc((size_t)n * 4);
  float*  pacc   = (float*)alloc((size_t)n * 4);
  float*  posw   = (float*)alloc((size_t)n * 4);
  __bf16* featbf = (__bf16*)alloc((size_t)n * INC * 2);
  __bf16* w1t    = (__bf16*)alloc((size_t)HID * INC * 2);
  __bf16* w2t    = (__bf16*)alloc((size_t)OUTC * HID * 2);
  float*  xw     = (float*)alloc((size_t)n * HID * 4);   // GEMM out (reused)
  float*  acc    = (float*)alloc((size_t)n * HID * 4);   // scatter accum (reused)
  float*  x1     = (float*)alloc((size_t)n * HID * 4);
  __bf16* x1bf   = (__bf16*)alloc((size_t)n * HID * 2);
  float*  x2     = (float*)alloc((size_t)n * OUTC * 4);
  __bf16* x2bf   = (__bf16*)alloc((size_t)n * OUTC * 2);
  float*  sim    = (float*)d_out;                        // [n,n], written by GEMM

  const int T = 256;
  auto cdiv = [](int a, int b) { return (a + b - 1) / b; };

  // ---- gcn_norm ----
  k_fill<<<cdiv(n, T), T, 0, stream>>>(deg, n, 1.0f);            // self-loops
  k_deg_accum<<<cdiv(e, T), T, 0, stream>>>(colp, deg, e);
  k_dinv<<<cdiv(n, T), T, 0, stream>>>(deg, dinv, n);

  // ---- weight prep ----
  k_cast_bf16<<<cdiv(n * INC, T), T, 0, stream>>>(feat, featbf, n * INC);
  k_transpose_cast<<<cdiv(INC * HID, T), T, 0, stream>>>(W1, w1t, INC, HID);
  k_transpose_cast<<<cdiv(HID * OUTC, T), T, 0, stream>>>(W2, w2t, HID, OUTC);

  // ---- layer 1: x1 = relu(conv(feat, W1)) ----
  k_gemm_bf16_nt_wmma<<<dim3(HID / 64, n / 128), T, 0, stream>>>(featbf, w1t, xw, n, HID, INC);
  k_fill<<<cdiv(n * HID, T), T, 0, stream>>>(acc, n * HID, 0.f);
  k_scatter_edges<<<e, T, 0, stream>>>(xw, rowp, colp, dinv, acc, HID);
  k_finalize<<<cdiv(n * HID, T), T, 0, stream>>>(acc, xw, dinv, b1, x1, n, HID, 1);
  k_cast_bf16<<<cdiv(n * HID, T), T, 0, stream>>>(x1, x1bf, n * HID);

  // ---- pos_w1 = sigmoid(conv(x1, W3)) ----
  k_dot_w3<<<n, T, 0, stream>>>(x1, W3, t3, HID);
  k_fill<<<cdiv(n, T), T, 0, stream>>>(pacc, n, 0.f);
  k_scatter_scalar<<<cdiv(e, T), T, 0, stream>>>(t3, rowp, colp, dinv, pacc, e);
  k_pos_final<<<cdiv(n, T), T, 0, stream>>>(pacc, t3, dinv, b3, posw, n);

  // ---- layer 2: x2 = conv(x1, W2) ----
  k_gemm_bf16_nt_wmma<<<dim3(OUTC / 64, n / 128), T, 0, stream>>>(x1bf, w2t, xw, n, OUTC, HID);
  k_fill<<<cdiv(n * OUTC, T), T, 0, stream>>>(acc, n * OUTC, 0.f);
  k_scatter_edges<<<e, T, 0, stream>>>(xw, rowp, colp, dinv, acc, OUTC);
  k_finalize<<<cdiv(n * OUTC, T), T, 0, stream>>>(acc, xw, dinv, b2, x2, n, OUTC, 0);
  k_cast_bf16<<<cdiv(n * OUTC, T), T, 0, stream>>>(x2, x2bf, n * OUTC);

  // ---- sim = x2 @ x2^T (B stored [N,K] == x2bf itself) ----
  k_gemm_bf16_nt_wmma<<<dim3(n / 64, n / 128), T, 0, stream>>>(x2bf, x2bf, sim, n, n, OUTC);

  // ---- row softmax * pos_w1, normalize, sigmoid (in-place on d_out) ----
  k_softmax_sigmoid<<<n, T, 0, stream>>>(sim, posw, n);
}
